// SingleMambaBlock_51496657879071
// MI455X (gfx1250) — matrix operand order repaired and checked by reading, not compiled, not run on hardware
//
#include <hip/hip_runtime.h>
#include <cstdint>
#include <cstddef>

// ---------------------------------------------------------------------------
// Mamba block for MI455X (gfx1250, wave32, WMMA).
// GEMMs: C[m,n] = sum_k A[m,k] * W[n,k], A/W f16 K-major, f32 accumulate via
// v_wmma_f32_16x16x32_f16. Operands are double-buffer staged into LDS with
// GLOBAL_LOAD_ASYNC_TO_LDS_B128 (ASYNCcnt) and consumed via ds_load_b128.
// ---------------------------------------------------------------------------

typedef __attribute__((ext_vector_type(16))) _Float16 v16h;
typedef __attribute__((ext_vector_type(8)))  _Float16 v8h;
typedef __attribute__((ext_vector_type(8)))  float    v8f;

#define MROWS  4096   // BATCH * SEQ
#define DMODEL 1024
#define DINNER 2048
#define E2     4096   // 2*DINNER
#define RANK   64
#define NST    16
#define SEQL   2048
#define DBLW   96     // RANK + 2*NST

// Block tile for the cooperative GEMM.
#define BM 128
#define BN 256
#define SLAB_A (BM * 32)            // 4096 f16  (8 KB)
#define SLAB_B (BN * 32)            // 8192 f16  (16 KB)
#define SLAB   (SLAB_A + SLAB_B)    // 12288 f16 (24 KB); x2 buffers = 48 KB LDS

// ------------------------- async global -> LDS copy -------------------------
// ISA: GLOBAL_LOAD_ASYNC_TO_LDS_B128 — per-lane 16B, tracked by ASYNCcnt.
// LDS destination offset = low 32 bits of the generic LDS pointer.
__device__ __forceinline__ void async_copy_b128(const _Float16* lds_dst,
                                                const _Float16* gsrc) {
  const uint32_t lds_off = (uint32_t)(uintptr_t)lds_dst;
  const unsigned long long g = (unsigned long long)(uintptr_t)gsrc;
  asm volatile("global_load_async_to_lds_b128 %0, %1, off"
               :: "v"(lds_off), "v"(g) : "memory");
}

__device__ __forceinline__ void wait_async_le6() {
  asm volatile("s_wait_asynccnt 0x6" ::: "memory");
}
__device__ __forceinline__ void wait_async_0() {
  asm volatile("s_wait_asynccnt 0x0" ::: "memory");
}

// ----------------------------- WMMA fragment loads --------------------------
// LDS slabs are row-major [rows x 32] f16 (64B rows).
// A frag (16x32): lanes 0-15 -> K 0..15, lanes 16-31 -> K 16..31.
__device__ __forceinline__ v16h lds_a_frag(const _Float16* sA, int m0l, int lane) {
  const int row  = m0l + (lane & 15);
  const int koff = (lane & 16) ? 16 : 0;
  const _Float16* p = sA + row * 32 + koff;
  v16h f;
  *reinterpret_cast<v8h*>(&f)       = *reinterpret_cast<const v8h*>(p);
  *(reinterpret_cast<v8h*>(&f) + 1) = *reinterpret_cast<const v8h*>(p + 8);
  return f;
}

// B frag (32x16): lane&15 = column N, lanes 0-15 hold K 0..15, 16-31 K 16..31.
__device__ __forceinline__ v16h lds_b_frag(const _Float16* sB, int n0l, int lane) {
  const int row  = n0l + (lane & 15);
  const int koff = (lane & 16) ? 16 : 0;
  return *reinterpret_cast<const v16h*>(sB + row * 32 + koff);
}

// -------------------------------- GEMM kernel -------------------------------
// 256 threads = 8 waves; block tile 128x256; wave tile 64x64 (4x4 v8f acc,
// 16 WMMAs per K=32 slab). A/B slabs double-buffered in LDS via async copies.
__global__ __launch_bounds__(256) void gemm_wmma_f16(const _Float16* __restrict__ A,
                                                     const _Float16* __restrict__ Bm,
                                                     float* __restrict__ C,
                                                     const float* __restrict__ resid,
                                                     int M, int N, int K) {
  __shared__ _Float16 smem[2 * SLAB];
  const int lane = threadIdx.x;                 // 0..31 (wave32)
  const int wid  = threadIdx.y;                 // 0..7
  const int t    = wid * 32 + lane;             // 0..255
  const int tilesN = (N + BN - 1) / BN;
  const int tmb = (blockIdx.x / tilesN) * BM;
  const int tnb = (blockIdx.x % tilesN) * BN;
  const int m0l = (wid >> 2) * 64;              // wave tile within block (2x4)
  const int n0l = (wid & 3) * 64;
  const int nk  = K >> 5;

  // ---- stage one K=32 slab into LDS buffer `buf` (6 async b128 / thread) ---
  auto stage = [&](int kk, int buf) {
    _Float16* sA = smem + buf * SLAB;
    _Float16* sB = sA + SLAB_A;
    const int k0 = kk << 5;
#pragma unroll
    for (int u = 0; u < 2; ++u) {               // A: 128x32 = 512 chunks
      const int c = t + u * 256;
      const int row = c >> 2, seg = c & 3;
      async_copy_b128(sA + row * 32 + seg * 8,
                      A + (size_t)(tmb + row) * K + k0 + seg * 8);
    }
#pragma unroll
    for (int u = 0; u < 4; ++u) {               // B: 256x32 = 1024 chunks
      const int c = t + u * 256;
      const int row = c >> 2, seg = c & 3;
      int n = tnb + row;
      if (n >= N) n = N - 1;                    // clamp (stores are masked)
      async_copy_b128(sB + row * 32 + seg * 8,
                      Bm + (size_t)n * K + k0 + seg * 8);
    }
  };

  const v8f zero = {0.f, 0.f, 0.f, 0.f, 0.f, 0.f, 0.f, 0.f};
  v8f acc[4][4];
#pragma unroll
  for (int i = 0; i < 4; ++i)
#pragma unroll
    for (int j = 0; j < 4; ++j) acc[i][j] = zero;

  stage(0, 0);
  for (int kk = 0; kk < nk; ++kk) {
    if (kk + 1 < nk) {
      stage(kk + 1, (kk + 1) & 1);
      wait_async_le6();                         // slab kk done (in-order), kk+1 in flight
    } else {
      wait_async_0();
    }
    __syncthreads();                            // slab kk visible to all waves

    const _Float16* sA = smem + (kk & 1) * SLAB;
    const _Float16* sB = sA + SLAB_A;
    v16h a[4], b[4];
#pragma unroll
    for (int i = 0; i < 4; ++i) a[i] = lds_a_frag(sA, m0l + 16 * i, lane);
#pragma unroll
    for (int j = 0; j < 4; ++j) b[j] = lds_b_frag(sB, n0l + 16 * j, lane);
#pragma unroll
    for (int i = 0; i < 4; ++i)
#pragma unroll
      for (int j = 0; j < 4; ++j)
        acc[i][j] = __builtin_amdgcn_wmma_f32_16x16x32_f16(
            false, a[i], false, b[j], (short)0, acc[i][j], false, false);

    __syncthreads();                            // reads done before buffer reuse
  }

  // Epilogue. C/D layout: VGPR e -> M = e + 8*(lane>=16), N = lane&15.
  const int lh = lane >> 4;
  const int lc = lane & 15;
#pragma unroll
  for (int i = 0; i < 4; ++i) {
#pragma unroll
    for (int j = 0; j < 4; ++j) {
      const int col = tnb + n0l + 16 * j + lc;
      if (col < N) {
#pragma unroll
        for (int e = 0; e < 8; ++e) {
          const int row = tmb + m0l + 16 * i + e + 8 * lh;
          const size_t idx = (size_t)row * N + col;
          float v = acc[i][j][e];
          if (resid) v += resid[idx];
          C[idx] = v;
        }
      }
    }
  }
}

// -------------------------------- LayerNorm ---------------------------------
__global__ __launch_bounds__(256) void layernorm_f16(const float* __restrict__ x,
                                                     const float* __restrict__ g,
                                                     const float* __restrict__ b,
                                                     _Float16* __restrict__ out) {
  const int row = blockIdx.x;                 // 0..MROWS-1
  const float* xr = x + (size_t)row * DMODEL;
  __shared__ float rs[256], rs2[256];
  float v[4], s = 0.f, s2 = 0.f;
#pragma unroll
  for (int i = 0; i < 4; ++i) {
    v[i] = xr[threadIdx.x + 256 * i];
    s += v[i];
    s2 += v[i] * v[i];
  }
  rs[threadIdx.x] = s;
  rs2[threadIdx.x] = s2;
  __syncthreads();
  for (int off = 128; off > 0; off >>= 1) {
    if (threadIdx.x < off) {
      rs[threadIdx.x]  += rs[threadIdx.x + off];
      rs2[threadIdx.x] += rs2[threadIdx.x + off];
    }
    __syncthreads();
  }
  const float mean = rs[0] * (1.f / DMODEL);
  const float var  = rs2[0] * (1.f / DMODEL) - mean * mean;
  const float rstd = rsqrtf(var + 1e-5f);
#pragma unroll
  for (int i = 0; i < 4; ++i) {
    const int c = threadIdx.x + 256 * i;
    out[(size_t)row * DMODEL + c] = (_Float16)((v[i] - mean) * rstd * g[c] + b[c]);
  }
}

// ------------------------ depthwise causal conv + SiLU ----------------------
__global__ __launch_bounds__(256) void conv_silu(const float* __restrict__ xz,
                                                 const float* __restrict__ w,
                                                 const float* __restrict__ bias,
                                                 float* __restrict__ xm_f32,
                                                 _Float16* __restrict__ xm_f16) {
  const int idx = blockIdx.x * 256 + threadIdx.x;   // over MROWS*DINNER
  if (idx >= MROWS * DINNER) return;
  const int d = idx & (DINNER - 1);
  const int m = idx >> 11;
  const int l = m & (SEQL - 1);
  float acc = bias[d];
#pragma unroll
  for (int k = 0; k < 4; ++k) {
    const int lk = l + k - 3;
    if (lk >= 0) acc += xz[(size_t)(m + k - 3) * E2 + d] * w[d * 4 + k];
  }
  const float sv = acc / (1.f + __expf(-acc));      // SiLU
  xm_f32[idx] = sv;
  xm_f16[idx] = (_Float16)sv;
}

// ---------------------------- small convert kernels -------------------------
__global__ void f32_to_f16(const float* __restrict__ in, _Float16* __restrict__ o, int n) {
  const int i = blockIdx.x * 256 + threadIdx.x;
  if (i < n) o[i] = (_Float16)in[i];
}

__global__ void slice_dt_f16(const float* __restrict__ dbl, _Float16* __restrict__ o) {
  const int i = blockIdx.x * 256 + threadIdx.x;     // MROWS*RANK
  if (i >= MROWS * RANK) return;
  const int m = i >> 6, r = i & 63;
  o[i] = (_Float16)dbl[(size_t)m * DBLW + r];
}

// ------------------------------ selective scan ------------------------------
// 4096 chains (b,d); block = 256 channels of one batch; B/C broadcast via LDS.
// Fuses softplus(dt), the SSM recurrence, D-skip, and SiLU(z) gating.
__global__ __launch_bounds__(256) void selective_scan(const float* __restrict__ dtlin,
                                                      const float* __restrict__ dt_b,
                                                      const float* __restrict__ dbl,
                                                      const float* __restrict__ xm,
                                                      const float* __restrict__ A_log,
                                                      const float* __restrict__ Dp,
                                                      const float* __restrict__ xz,
                                                      _Float16* __restrict__ ygate) {
  const int b = blockIdx.x >> 3;                    // 2 batches * 8 blocks
  const int d = ((blockIdx.x & 7) << 8) + threadIdx.x;
  float Ar[NST];
#pragma unroll
  for (int n = 0; n < NST; ++n) Ar[n] = -__expf(A_log[d * NST + n]);
  const float bdt = dt_b[d];
  const float Dd  = Dp[d];
  float h[NST];
#pragma unroll
  for (int n = 0; n < NST; ++n) h[n] = 0.f;

  __shared__ float sBC[2 * NST];                    // [0..15]=B_t, [16..31]=C_t
  for (int l = 0; l < SEQL; ++l) {
    const size_t m = (size_t)b * SEQL + l;
    if (threadIdx.x < 2 * NST) sBC[threadIdx.x] = dbl[m * DBLW + RANK + threadIdx.x];
    __syncthreads();

    const float dv   = dtlin[m * DINNER + d] + bdt;
    const float dt_t = (dv > 20.f) ? dv : log1pf(__expf(dv));   // softplus
    const float x_t  = xm[m * DINNER + d];
    const float dtx  = dt_t * x_t;
    float y = 0.f;
#pragma unroll
    for (int n = 0; n < NST; ++n) {
      const float dA = __expf(dt_t * Ar[n]);
      h[n] = dA * h[n] + dtx * sBC[n];
      y += h[n] * sBC[NST + n];
    }
    y += Dd * x_t;
    const float z = xz[m * E2 + DINNER + d];
    y *= z / (1.f + __expf(-z));                    // * SiLU(z)
    ygate[m * DINNER + d] = (_Float16)y;
    __syncthreads();
  }
}

// --------------------------------- launcher ---------------------------------
extern "C" void kernel_launch(void* const* d_in, const int* in_sizes, int n_in,
                              void* d_out, int out_size, void* d_ws, size_t ws_size,
                              hipStream_t stream) {
  const float* x        = (const float*)d_in[0];   // (2,2048,1024)
  const float* ln_g     = (const float*)d_in[1];
  const float* ln_b     = (const float*)d_in[2];
  const float* in_projw = (const float*)d_in[3];   // (4096,1024)
  const float* conv_w   = (const float*)d_in[4];   // (2048,1,4)
  const float* conv_b   = (const float*)d_in[5];
  const float* x_projw  = (const float*)d_in[6];   // (96,2048)
  const float* dt_projw = (const float*)d_in[7];   // (2048,64)
  const float* dt_projb = (const float*)d_in[8];
  const float* A_log    = (const float*)d_in[9];   // (2048,16)
  const float* Dp       = (const float*)d_in[10];
  const float* out_projw= (const float*)d_in[11];  // (1024,2048)
  float* out = (float*)d_out;

  // ----- workspace carve-out -----
  char* p = (char*)d_ws;
  auto carve = [&](size_t bytes) -> char* {
    char* r = p;
    p += (bytes + 255) & ~(size_t)255;
    return r;
  };
  _Float16* xn16   = (_Float16*)carve((size_t)MROWS * DMODEL * 2);
  _Float16* win16  = (_Float16*)carve((size_t)E2 * DMODEL * 2);
  _Float16* wxp16  = (_Float16*)carve((size_t)DBLW * DINNER * 2);
  _Float16* wdt16  = (_Float16*)carve((size_t)DINNER * RANK * 2);
  _Float16* wout16 = (_Float16*)carve((size_t)DMODEL * DINNER * 2);
  float*    xz     = (float*)carve((size_t)MROWS * E2 * 4);
  float*    xmf    = (float*)carve((size_t)MROWS * DINNER * 4);
  _Float16* xm16   = (_Float16*)carve((size_t)MROWS * DINNER * 2);
  float*    dbl    = (float*)carve((size_t)MROWS * DBLW * 4);
  _Float16* dtin16 = (_Float16*)carve((size_t)MROWS * RANK * 2);
  float*    dtlin  = (float*)carve((size_t)MROWS * DINNER * 4);
  _Float16* yg16   = (_Float16*)carve((size_t)MROWS * DINNER * 2);

  const dim3 blk256(256);
  const dim3 gblk(32, 8);
  auto gemm_grid = [](int M, int N) -> dim3 {
    return dim3((M / BM) * ((N + BN - 1) / BN));
  };

  // 1) LayerNorm -> xn (f16)
  layernorm_f16<<<MROWS, blk256, 0, stream>>>(x, ln_g, ln_b, xn16);

  // 2) weight conversions f32 -> f16
  f32_to_f16<<<(E2 * DMODEL + 255) / 256, blk256, 0, stream>>>(in_projw, win16, E2 * DMODEL);
  f32_to_f16<<<(DBLW * DINNER + 255) / 256, blk256, 0, stream>>>(x_projw, wxp16, DBLW * DINNER);
  f32_to_f16<<<(DINNER * RANK + 255) / 256, blk256, 0, stream>>>(dt_projw, wdt16, DINNER * RANK);
  f32_to_f16<<<(DMODEL * DINNER + 255) / 256, blk256, 0, stream>>>(out_projw, wout16, DMODEL * DINNER);

  // 3) in_proj: xz[4096,4096] = xn[4096,1024] * Win^T
  gemm_wmma_f16<<<gemm_grid(MROWS, E2), gblk, 0, stream>>>(
      xn16, win16, xz, nullptr, MROWS, E2, DMODEL);

  // 4) causal depthwise conv + SiLU on x-half of xz
  conv_silu<<<(MROWS * DINNER + 255) / 256, blk256, 0, stream>>>(
      xz, conv_w, conv_b, xmf, xm16);

  // 5) x_proj: dbl[4096,96] = xm[4096,2048] * Wxp^T
  gemm_wmma_f16<<<gemm_grid(MROWS, DBLW), gblk, 0, stream>>>(
      xm16, wxp16, dbl, nullptr, MROWS, DBLW, DINNER);

  // 6) dt low-rank input slice -> f16
  slice_dt_f16<<<(MROWS * RANK + 255) / 256, blk256, 0, stream>>>(dbl, dtin16);

  // 7) dt_proj: dtlin[4096,2048] = dtin[4096,64] * Wdt^T
  gemm_wmma_f16<<<gemm_grid(MROWS, DINNER), gblk, 0, stream>>>(
      dtin16, wdt16, dtlin, nullptr, MROWS, DINNER, RANK);

  // 8) selective scan (+softplus, D-skip, SiLU(z) gate) -> y (f16)
  selective_scan<<<16, blk256, 0, stream>>>(
      dtlin, dt_projb, dbl, xmf, A_log, Dp, xz, yg16);

  // 9) out_proj + residual: out[4096,1024] = y[4096,2048]*Wout^T + x
  gemm_wmma_f16<<<gemm_grid(MROWS, DMODEL), gblk, 0, stream>>>(
      yg16, wout16, out, x, MROWS, DMODEL, DINNER);
}
